// PrimeKGDrugRepurposingGNN_12120397709965
// MI455X (gfx1250) — matrix operand order repaired and checked by reading, not compiled, not run on hardware
//
#include <hip/hip_runtime.h>
#include <hip/hip_bf16.h>

#define LN_EPS 1e-5f

typedef __attribute__((ext_vector_type(16))) __bf16 bf16x16;
typedef __attribute__((ext_vector_type(8)))  float  floatx8;

union Frag32B { bf16x16 v; uint4 u[2]; };

// ---------------- encode: x = node_emb + type_emb[type] ----------------
__global__ __launch_bounds__(256)
void k_encode(const float* __restrict__ node_emb, const float* __restrict__ type_emb,
              const int* __restrict__ type_ids, float* __restrict__ x,
              int N, int H) {
  size_t i = (size_t)blockIdx.x * blockDim.x + threadIdx.x;
  if (i >= (size_t)N * H) return;
  int n = (int)(i / H), f = (int)(i % H);
  x[i] = node_emb[i] + type_emb[(size_t)type_ids[n] * H + f];
}

__global__ __launch_bounds__(256)
void k_zero(float* __restrict__ p, size_t n) {
  size_t i = (size_t)blockIdx.x * blockDim.x + threadIdx.x;
  if (i < n) p[i] = 0.f;
}

// ------- spmm: out[row] += w * x[col]; 4 edges / block, float4 gathers, f32 atomics -------
__global__ __launch_bounds__(256)
void k_spmm4(const int* __restrict__ row, const int* __restrict__ col,
             const float* __restrict__ w, const float* __restrict__ x,
             float* __restrict__ out, int H, int NE) {
  int e = blockIdx.x * 4 + (threadIdx.x >> 6);
  if (e >= NE) return;
  int f = (threadIdx.x & 63) * 4;            // H == 256 == 64 lanes * 4 floats
  int r = row[e], c = col[e];
  float we = w[e];
  float4 xv = *(const float4*)(x + (size_t)c * H + f);
  float* o = out + (size_t)r * H + f;
  atomicAdd(o + 0, we * xv.x);
  atomicAdd(o + 1, we * xv.y);
  atomicAdd(o + 2, we * xv.z);
  atomicAdd(o + 3, we * xv.w);
}

// ---------------- fp32 -> bf16, 4-wide ----------------
__global__ __launch_bounds__(256)
void k_f32_to_bf16v(const float* __restrict__ src, __bf16* __restrict__ dst, size_t n4) {
  size_t i = (size_t)blockIdx.x * blockDim.x + threadIdx.x;
  if (i >= n4) return;
  float4 v = ((const float4*)src)[i];
  union { __bf16 h[4]; uint2 u; } o;
  o.h[0] = (__bf16)v.x; o.h[1] = (__bf16)v.y;
  o.h[2] = (__bf16)v.z; o.h[3] = (__bf16)v.w;
  ((uint2*)dst)[i] = o.u;
}

// -------- weight transpose (+K pad) to bf16 : Wt[n*Kpad+k] = W[k*N+n] --------
__global__ __launch_bounds__(256)
void k_wt_bf16(const float* __restrict__ W, __bf16* __restrict__ Wt,
               int K, int N, int Kpad) {
  size_t i = (size_t)blockIdx.x * blockDim.x + threadIdx.x;
  if (i >= (size_t)N * Kpad) return;
  int n = (int)(i / Kpad), k = (int)(i % Kpad);
  Wt[i] = (k < K) ? (__bf16)W[(size_t)k * N + n] : (__bf16)0.f;
}

// ---- WMMA GEMM: C[M,N] = A[M,K](bf16) @ Bt[N,K](bf16)^T + bias, opt relu ----
// register blocking: each wave computes a 32(M) x 64(N) patch = 2x4 wmma tiles.
// grid.x covers M/32 (4 waves/block), grid.y covers N/64. M%32==0, N%64==0, K%32==0.
#define MT 2
#define NT 4
__global__ __launch_bounds__(128)
void k_gemm_bf16(const __bf16* __restrict__ A, const __bf16* __restrict__ Bt,
                 const float* __restrict__ bias, float* __restrict__ C,
                 int M, int N, int K, int relu) {
  int wid  = threadIdx.x >> 5;
  int lane = threadIdx.x & 31;
  int mblk = blockIdx.x * 4 + wid;
  if (mblk * 32 >= M) return;                // wave-uniform: EXEC all-1s inside
  int m0   = mblk * 32;
  int n0   = blockIdx.y * (NT * 16);
  int half = lane >> 4;                      // K half select: contiguous order k = 16*half + s in A and B
  int l16  = lane & 15;

  const __bf16* Ar[MT];
#pragma unroll
  for (int mt = 0; mt < MT; ++mt) Ar[mt] = A + (size_t)(m0 + mt * 16 + l16) * K;
  const __bf16* Bc[NT];
#pragma unroll
  for (int nt = 0; nt < NT; ++nt)
    Bc[nt] = Bt + (size_t)(n0 + nt * 16 + l16) * K + half * 16;

  floatx8 acc[MT][NT] = {};
  for (int k0 = 0; k0 < K; k0 += 32) {
    Frag32B a[MT];
#pragma unroll
    for (int mt = 0; mt < MT; ++mt) {
      a[mt].u[0] = *(const uint4*)(Ar[mt] + k0 + half * 16);
      a[mt].u[1] = *(const uint4*)(Ar[mt] + k0 + half * 16 + 8);
      __builtin_prefetch(Ar[mt] + k0 + 32 + half * 8, 0, 0);   // next K-step of A stream
    }
#pragma unroll
    for (int nt = 0; nt < NT; ++nt) {
      Frag32B b;
      b.u[0] = *(const uint4*)(Bc[nt] + k0);
      b.u[1] = *(const uint4*)(Bc[nt] + k0 + 8);
#pragma unroll
      for (int mt = 0; mt < MT; ++mt)
        acc[mt][nt] = __builtin_amdgcn_wmma_f32_16x16x32_bf16(
            false, a[mt].v, false, b.v, (short)0, acc[mt][nt], false, false);
    }
  }

#pragma unroll
  for (int mt = 0; mt < MT; ++mt) {
#pragma unroll
    for (int nt = 0; nt < NT; ++nt) {
      float bn = bias[n0 + nt * 16 + l16];
      float* Crow = C + (size_t)(m0 + mt * 16 + half * 8) * N + n0 + nt * 16 + l16;
#pragma unroll
      for (int r = 0; r < 8; ++r) {
        float v = acc[mt][nt][r] + bn;
        if (relu) v = fmaxf(v, 0.f);
        Crow[(size_t)r * N] = v;
      }
    }
  }
}

// ---------------- x += relu(layer_norm(h)) , one block (256 thr) per row, H==256 ----------
__global__ __launch_bounds__(256)
void k_ln_relu_res(float* __restrict__ x, const float* __restrict__ h,
                   const float* __restrict__ g, const float* __restrict__ be, int H) {
  __shared__ float red[256];
  int row = blockIdx.x, t = threadIdx.x;
  float v = h[(size_t)row * H + t];
  red[t] = v; __syncthreads();
  for (int s = 128; s > 0; s >>= 1) { if (t < s) red[t] += red[t + s]; __syncthreads(); }
  float mu = red[0] / H; __syncthreads();
  float d = v - mu;
  red[t] = d * d; __syncthreads();
  for (int s = 128; s > 0; s >>= 1) { if (t < s) red[t] += red[t + s]; __syncthreads(); }
  float var = red[0] / H;
  float y = g[t] * d * rsqrtf(var + LN_EPS) + be[t];
  x[(size_t)row * H + t] += fmaxf(y, 0.f);
}

// ---------------- build scorer features, bf16, K padded 386 -> 416 ----------------
__global__ __launch_bounds__(256)
void k_build_feat(const float* __restrict__ z, const int* __restrict__ pairs,
                  const int* __restrict__ deg, __bf16* __restrict__ feat,
                  int P, int ED, int Kpad) {
  int p = blockIdx.x, t = threadIdx.x;
  int src = pairs[p], dst = pairs[P + p];
  __bf16* fr = feat + (size_t)p * Kpad;
  if (t < ED) {
    float s = z[(size_t)src * ED + t];
    float d = z[(size_t)dst * ED + t];
    fr[t]          = (__bf16)s;
    fr[ED + t]     = (__bf16)d;
    fr[2 * ED + t] = (__bf16)(s * d);
  } else {
    int c = 3 * ED + (t - ED);               // covers [384, 512)
    if (c < Kpad) {
      float v = 0.f;
      if (c == 3 * ED) { int dg = deg[src]; v = logf((float)(dg > 1 ? dg : 1)); }
      else if (c == 3 * ED + 1) { int dg = deg[dst]; v = logf((float)(dg > 1 ? dg : 1)); }
      fr[c] = (__bf16)v;
    }
  }
}

// ---------------- batch-norm stats per column: stats[c]=mu, stats[ED+c]=var --------------
__global__ __launch_bounds__(256)
void k_bn_stats(const float* __restrict__ hp, float* __restrict__ stats, int P, int ED) {
  __shared__ float s1[256], s2[256];
  int c = blockIdx.x, t = threadIdx.x;
  float a = 0.f, b = 0.f;
  for (int p = t; p < P; p += 256) { float v = hp[(size_t)p * ED + c]; a += v; b += v * v; }
  s1[t] = a; s2[t] = b; __syncthreads();
  for (int s = 128; s > 0; s >>= 1) {
    if (t < s) { s1[t] += s1[t + s]; s2[t] += s2[t + s]; }
    __syncthreads();
  }
  if (t == 0) { float mu = s1[0] / P; stats[c] = mu; stats[ED + c] = s2[0] / P - mu * mu; }
}

// ---------------- final: out[p] = bn(hp[p]) . W_l2 + b_l2 ; one wave32 per pair -----------
__global__ __launch_bounds__(256)
void k_score(const float* __restrict__ hp, const float* __restrict__ stats,
             const float* __restrict__ bn_g, const float* __restrict__ bn_b,
             const float* __restrict__ W2, const float* __restrict__ b2,
             float* __restrict__ out, int P, int ED) {
  int wid = threadIdx.x >> 5, lane = threadIdx.x & 31;
  int p = blockIdx.x * 8 + wid;
  if (p >= P) return;
  float acc = 0.f;
  for (int c = lane; c < ED; c += 32) {
    float yn = bn_g[c] * (hp[(size_t)p * ED + c] - stats[c]) * rsqrtf(stats[ED + c] + LN_EPS)
             + bn_b[c];
    acc += yn * W2[c];
  }
  for (int m = 16; m > 0; m >>= 1) acc += __shfl_xor(acc, m, 32);
  if (lane == 0) out[p] = acc + b2[0];
}

extern "C" void kernel_launch(void* const* d_in, const int* in_sizes, int n_in,
                              void* d_out, int out_size, void* d_ws, size_t ws_size,
                              hipStream_t stream) {
  const int*   type_ids = (const int*)d_in[0];
  const int*   erow     = (const int*)d_in[1];
  const int*   ecol     = (const int*)d_in[2];
  const float* ew       = (const float*)d_in[3];
  const int*   pairs    = (const int*)d_in[4];
  const int*   degrees  = (const int*)d_in[5];
  const float* node_emb = (const float*)d_in[6];
  const float* type_emb = (const float*)d_in[7];
  const float* W_in  = (const float*)d_in[8];   const float* b_in  = (const float*)d_in[9];
  const float* W_r1  = (const float*)d_in[10];  const float* b_r1  = (const float*)d_in[11];
  const float* g_r1  = (const float*)d_in[12];  const float* be_r1 = (const float*)d_in[13];
  const float* W_r2  = (const float*)d_in[14];  const float* b_r2  = (const float*)d_in[15];
  const float* g_r2  = (const float*)d_in[16];  const float* be_r2 = (const float*)d_in[17];
  const float* W_out = (const float*)d_in[18];  const float* b_out = (const float*)d_in[19];
  const float* W_l1  = (const float*)d_in[20];  const float* b_l1  = (const float*)d_in[21];
  const float* bn_g  = (const float*)d_in[22];  const float* bn_b  = (const float*)d_in[23];
  const float* W_l2  = (const float*)d_in[24];  const float* b_l2  = (const float*)d_in[25];

  const int N  = in_sizes[0];
  const int NE = in_sizes[1];
  const int P  = in_sizes[4] / 2;
  const int H  = in_sizes[9];                  // 256
  const int ED = in_sizes[19];                 // 128
  const int KP = 416;                          // scorer K padded: 3*ED+2=386 -> 13*32

  const size_t NH = (size_t)N * H;

  // ---- workspace layout ----
  float*  xf   = (float*)d_ws;                 // [N,H] current features
  float*  ax   = xf + NH;                      // [N,H] spmm accumulator
  float*  hf   = ax + NH;                      // [N,H] gemm temp
  __bf16* Ab   = (__bf16*)(hf + NH);           // [N,H] bf16 A
  float*  zf   = (float*)(Ab + NH);            // [N,ED]
  __bf16* Wb   = (__bf16*)(zf + (size_t)N * ED);    // [H,H] transposed bf16 weight
  __bf16* Wl1b = Wb + (size_t)H * H;                // [128,416]
  float*  stats = (float*)(Wl1b + (size_t)KP * ED); // mu[ED], var[ED]
  __bf16* featb = (__bf16*)ax;                 // alias: ax dead after last conversion
  float*  hp    = hf;                          // alias: hf dead after r2 layernorm
  float*  outp  = (float*)d_out;

  dim3 b256(256), b128(128);
  unsigned gNH   = (unsigned)((NH + 255) / 256);
  unsigned gNH4  = (unsigned)((NH / 4 + 255) / 256);
  unsigned gSp   = (unsigned)((NE + 3) / 4);
  unsigned gGx   = (unsigned)((N / 32 + 3) / 4);
  unsigned gGxP  = (unsigned)((P / 32 + 3) / 4);
  unsigned gW_HH = (unsigned)(((size_t)H * H + 255) / 256);
  unsigned gW_HE = (unsigned)(((size_t)H * ED + 255) / 256);
  unsigned gWl1  = (unsigned)(((size_t)ED * KP + 255) / 256);
  dim3 gridH(gGx, H / 64);                     // GEMM -> [N,H]
  dim3 gridE(gGx, ED / 64);                    // GEMM -> [N,ED]
  dim3 gridS(gGxP, ED / 64);                   // scorer GEMM -> [P,ED]

  // ===== encode =====
  k_encode<<<gNH, b256, 0, stream>>>(node_emb, type_emb, type_ids, xf, N, H);

  // ===== layer in: x = relu(spmm(x) @ W_in + b_in) =====
  k_zero<<<gNH, b256, 0, stream>>>(ax, NH);
  k_spmm4<<<gSp, b256, 0, stream>>>(erow, ecol, ew, xf, ax, H, NE);
  k_f32_to_bf16v<<<gNH4, b256, 0, stream>>>(ax, Ab, NH / 4);
  k_wt_bf16<<<gW_HH, b256, 0, stream>>>(W_in, Wb, H, H, H);
  k_gemm_bf16<<<gridH, b128, 0, stream>>>(Ab, Wb, b_in, xf, N, H, H, 1);

  // ===== residual layer 1 =====
  k_zero<<<gNH, b256, 0, stream>>>(ax, NH);
  k_spmm4<<<gSp, b256, 0, stream>>>(erow, ecol, ew, xf, ax, H, NE);
  k_f32_to_bf16v<<<gNH4, b256, 0, stream>>>(ax, Ab, NH / 4);
  k_wt_bf16<<<gW_HH, b256, 0, stream>>>(W_r1, Wb, H, H, H);
  k_gemm_bf16<<<gridH, b128, 0, stream>>>(Ab, Wb, b_r1, hf, N, H, H, 0);
  k_ln_relu_res<<<N, b256, 0, stream>>>(xf, hf, g_r1, be_r1, H);

  // ===== residual layer 2 =====
  k_zero<<<gNH, b256, 0, stream>>>(ax, NH);
  k_spmm4<<<gSp, b256, 0, stream>>>(erow, ecol, ew, xf, ax, H, NE);
  k_f32_to_bf16v<<<gNH4, b256, 0, stream>>>(ax, Ab, NH / 4);
  k_wt_bf16<<<gW_HH, b256, 0, stream>>>(W_r2, Wb, H, H, H);
  k_gemm_bf16<<<gridH, b128, 0, stream>>>(Ab, Wb, b_r2, hf, N, H, H, 0);
  k_ln_relu_res<<<N, b256, 0, stream>>>(xf, hf, g_r2, be_r2, H);

  // ===== output projection: z = spmm(x) @ W_out + b_out =====
  k_zero<<<gNH, b256, 0, stream>>>(ax, NH);
  k_spmm4<<<gSp, b256, 0, stream>>>(erow, ecol, ew, xf, ax, H, NE);
  k_f32_to_bf16v<<<gNH4, b256, 0, stream>>>(ax, Ab, NH / 4);
  k_wt_bf16<<<gW_HE, b256, 0, stream>>>(W_out, Wb, H, ED, H);
  k_gemm_bf16<<<gridE, b128, 0, stream>>>(Ab, Wb, b_out, zf, N, ED, H, 0);

  // ===== scorer =====
  k_wt_bf16<<<gWl1, b256, 0, stream>>>(W_l1, Wl1b, 3 * ED + 2, ED, KP);
  k_build_feat<<<P, b256, 0, stream>>>(zf, pairs, degrees, featb, P, ED, KP);
  k_gemm_bf16<<<gridS, b128, 0, stream>>>(featb, Wl1b, b_l1, hp, P, ED, KP, 1);
  k_bn_stats<<<ED, b256, 0, stream>>>(hp, stats, P, ED);
  k_score<<<(unsigned)((P + 7) / 8), b256, 0, stream>>>(hp, stats, bn_g, bn_b, W_l2, b_l2,
                                                        outp, P, ED);
}
